// PhysGNN_46059229282812
// MI455X (gfx1250) — compile-verified, matrix operand order checked by reading, not run on hardware
//
#include <hip/hip_runtime.h>
#include <hip/hip_bf16.h>
#include <math.h>

// ---------------------------------------------------------------------------
// Types for CDNA5 WMMA
// ---------------------------------------------------------------------------
typedef __attribute__((ext_vector_type(16))) __bf16 bf16x16;
typedef __attribute__((ext_vector_type(8)))  float  f32x8;

#define N_NODES 50000
#define N_EDGES 800000
#define D_IN    14
#define HID     112
#define LSTM_H  168
#define G4      (4 * LSTM_H)   // 672

enum { GF_ACC = 1, GF_PRELU = 2 };

__device__ __forceinline__ __bf16 f32_to_bf16(float f) {
    union { float f; unsigned u; } v; v.f = f;
    unsigned r = v.u + 0x7FFFu + ((v.u >> 16) & 1u);   // round-to-nearest-even
    unsigned short h = (unsigned short)(r >> 16);
    __bf16 b; __builtin_memcpy(&b, &h, 2); return b;
}
__device__ __forceinline__ float sigf(float x) { return 1.0f / (1.0f + __expf(-x)); }

// ---------------------------------------------------------------------------
// Small utility kernels
// ---------------------------------------------------------------------------
__global__ void fill_kernel(float* __restrict__ p, float v, long n) {
    long i = blockIdx.x * (long)blockDim.x + threadIdx.x;
    if (i < n) p[i] = v;
}

__global__ void vec_add_kernel(const float* __restrict__ a, const float* __restrict__ b,
                               float* __restrict__ o, int n) {
    int i = blockIdx.x * blockDim.x + threadIdx.x;
    if (i < n) o[i] = a[i] + b[i];
}

// Stage f32 activations [rows x C] (stride C) into bf16 [rows x KP], zero-padded.
__global__ void cvt_bf16_kernel(const float* __restrict__ src, __bf16* __restrict__ dst,
                                int C, int KP, long rows) {
    long tid = blockIdx.x * (long)blockDim.x + threadIdx.x;
    if (tid >= rows * KP) return;
    long r = tid / KP; int c = (int)(tid - r * KP);
    dst[tid] = (c < C) ? f32_to_bf16(src[r * C + c]) : f32_to_bf16(0.0f);
}

// Convert f32 weights into bf16 [Np x Kp] ("Bt") layout, zero-padded.
// transpose==1: src is [K x N] (in,out) ; transpose==0: src already [N x K]
__global__ void prep_weight_kernel(const float* __restrict__ W, __bf16* __restrict__ Bt,
                                   int K, int N, int Kp, int Np, int transpose) {
    long idx = blockIdx.x * (long)blockDim.x + threadIdx.x;
    if (idx >= (long)Kp * Np) return;
    int n = (int)(idx / Kp), k = (int)(idx - (long)n * Kp);
    float v = 0.0f;
    if (n < N && k < K) v = transpose ? W[(long)k * N + n] : W[(long)n * K + k];
    Bt[idx] = f32_to_bf16(v);
}

// ---------------------------------------------------------------------------
// Edge scatter kernels (the HBM-bound part: coalesced along channels)
// ---------------------------------------------------------------------------
__global__ void scatter_sum_kernel(const float* __restrict__ x, const int* __restrict__ src,
                                   const int* __restrict__ dst, const float* __restrict__ ea,
                                   float* __restrict__ agg, long E, int C) {
    long tid = blockIdx.x * (long)blockDim.x + threadIdx.x;
    if (tid >= E * C) return;
    long e = tid / C; int c = (int)(tid - e * C);
    float w = 1.0f / ea[e];
    atomicAdd(&agg[(long)dst[e] * C + c], x[(long)src[e] * C + c] * w);
}

__device__ __forceinline__ void atomicMaxF(float* addr, float val) {
    if (val >= 0.0f) atomicMax((int*)addr, __float_as_int(val));
    else             atomicMin((unsigned int*)addr, __float_as_uint(val));
}

__global__ void scatter_max_kernel(const float* __restrict__ x, const int* __restrict__ src,
                                   const int* __restrict__ dst, const float* __restrict__ ea,
                                   float* __restrict__ agg, long E, int C, int use_ew) {
    long tid = blockIdx.x * (long)blockDim.x + threadIdx.x;
    if (tid >= E * C) return;
    long e = tid / C; int c = (int)(tid - e * C);
    float v = x[(long)src[e] * C + c];
    if (use_ew) v *= 1.0f / ea[e];
    atomicMaxF(&agg[(long)dst[e] * C + c], v);
}

__global__ void finalize_max_kernel(float* __restrict__ agg, long n) {
    long i = blockIdx.x * (long)blockDim.x + threadIdx.x;
    if (i >= n) return;
    float v = agg[i];
    if (!__builtin_isfinite(v)) agg[i] = 0.0f;
}

// ---------------------------------------------------------------------------
// bf16 WMMA GEMM: D[M x N] = A @ B (+ bias | + D) (opt PReLU)
//   A: bf16 [M x KP] zero-padded.  Bt: bf16 [Np x KP] zero-padded (Np mult 64).
//   One wave -> 16 rows x 64 cols (4 n-tiles, A-fragment reuse).
//   KP compile-time -> fully unrolled guard-free K loop, pure b128 loads + wmma.
// ---------------------------------------------------------------------------
template <int KP>
__global__ __launch_bounds__(128)
void gemm_wmma_kernel(const __bf16* __restrict__ A, const __bf16* __restrict__ Bt,
                      const float* __restrict__ bias, const float* __restrict__ pa,
                      float* __restrict__ D, int ldd, int M, int N, int flags)
{
    const int lane   = threadIdx.x & 31;
    const int wid    = threadIdx.x >> 5;
    const int mtile  = blockIdx.x * 4 + wid;
    if (mtile * 16 >= M) return;                 // wave-uniform
    const int half   = lane >> 4;                // 0 or 1
    const int ncol   = lane & 15;
    const int ntile0 = blockIdx.y * 4;

    f32x8 acc[4];
    #pragma unroll
    for (int t = 0; t < 4; ++t) {
        int n = (ntile0 + t) * 16 + ncol;
        bool ok = n < N;
        #pragma unroll
        for (int r = 0; r < 8; ++r) {
            float v = 0.0f;
            if (ok) {
                int m = mtile * 16 + r + 8 * half;
                if (flags & GF_ACC) v = D[(long)m * ldd + n];
                else if (bias)      v = bias[n];
            }
            acc[t][r] = v;
        }
    }

    // Lane's K-runs: [kb + half*8, +8) and [kb + 16 + half*8, +8)  (bf16 -> 16B each)
    const __bf16* Arow = A + (long)(mtile * 16 + ncol) * KP + half * 8;
    const __bf16* Brow0 = Bt + (long)((ntile0 + 0) * 16 + ncol) * KP + half * 8;
    const __bf16* Brow1 = Bt + (long)((ntile0 + 1) * 16 + ncol) * KP + half * 8;
    const __bf16* Brow2 = Bt + (long)((ntile0 + 2) * 16 + ncol) * KP + half * 8;
    const __bf16* Brow3 = Bt + (long)((ntile0 + 3) * 16 + ncol) * KP + half * 8;
    const __bf16* Brows[4] = { Brow0, Brow1, Brow2, Brow3 };

    #pragma unroll
    for (int kb = 0; kb < KP; kb += 32) {
        union { bf16x16 v; char b[32]; } fa;
        __builtin_memcpy(&fa.b[0],  Arow + kb,      16);
        __builtin_memcpy(&fa.b[16], Arow + kb + 16, 16);
        #pragma unroll
        for (int t = 0; t < 4; ++t) {
            union { bf16x16 v; char b[32]; } fb;
            __builtin_memcpy(&fb.b[0],  Brows[t] + kb,      16);
            __builtin_memcpy(&fb.b[16], Brows[t] + kb + 16, 16);
            acc[t] = __builtin_amdgcn_wmma_f32_16x16x32_bf16(
                false, fa.v, false, fb.v, (short)0, acc[t], false, false);
        }
    }

    #pragma unroll
    for (int t = 0; t < 4; ++t) {
        int n = (ntile0 + t) * 16 + ncol;
        if (n >= N) continue;
        float av = (flags & GF_PRELU) ? pa[n] : 0.0f;
        #pragma unroll
        for (int r = 0; r < 8; ++r) {
            int m = mtile * 16 + r + 8 * half;
            float v = acc[t][r];
            if (flags & GF_PRELU) v = (v >= 0.0f) ? v : av * v;
            D[(long)m * ldd + n] = v;
        }
    }
}

// ---------------------------------------------------------------------------
// LSTM pointwise: gates (N x 672) -> c,h update; stash h into hall[:, t, :]
// ---------------------------------------------------------------------------
__global__ void lstm_point_kernel(const float* __restrict__ g, float* __restrict__ c,
                                  float* __restrict__ hcur, float* __restrict__ hall,
                                  int t, int Nn)
{
    long tid = blockIdx.x * (long)blockDim.x + threadIdx.x;
    if (tid >= (long)Nn * LSTM_H) return;
    long node = tid / LSTM_H; int j = (int)(tid - node * LSTM_H);
    long base = node * G4;
    float iv = sigf(g[base + j]);
    float fv = sigf(g[base + LSTM_H + j]);
    float gv = tanhf(g[base + 2 * LSTM_H + j]);
    float ov = sigf(g[base + 3 * LSTM_H + j]);
    float cc = fv * c[tid] + iv * gv;
    float hh = ov * tanhf(cc);
    c[tid] = cc;
    hcur[tid] = hh;
    hall[node * (3 * LSTM_H) + (long)t * LSTM_H + j] = hh;
}

// ---------------------------------------------------------------------------
// JK attention: scores over 3 timesteps from [hf|hb], softmax, mix x1..x3.
// One wave per node.
// ---------------------------------------------------------------------------
__global__ __launch_bounds__(128)
void jk_attn_kernel(const float* __restrict__ x1, const float* __restrict__ x2,
                    const float* __restrict__ x3, const float* __restrict__ hf,
                    const float* __restrict__ hb, const float* __restrict__ Watt,
                    const float* __restrict__ batt, float* __restrict__ out, int Nn)
{
    int node = blockIdx.x * (blockDim.x >> 5) + (threadIdx.x >> 5);
    int lane = threadIdx.x & 31;
    if (node >= Nn) return;
    float s[3];
    #pragma unroll
    for (int t = 0; t < 3; ++t) {
        float acc = 0.0f;
        long base = (long)node * (3 * LSTM_H) + (long)t * LSTM_H;
        for (int d = lane; d < 2 * LSTM_H; d += 32) {
            float hv = (d < LSTM_H) ? hf[base + d] : hb[base + (d - LSTM_H)];
            acc += hv * Watt[d];
        }
        #pragma unroll
        for (int off = 16; off > 0; off >>= 1) acc += __shfl_xor(acc, off, 32);
        s[t] = acc + batt[0];
    }
    float mx = fmaxf(s[0], fmaxf(s[1], s[2]));
    float e0 = __expf(s[0] - mx), e1 = __expf(s[1] - mx), e2 = __expf(s[2] - mx);
    float inv = 1.0f / (e0 + e1 + e2);
    float a0 = e0 * inv, a1 = e1 * inv, a2 = e2 * inv;
    long rb = (long)node * HID;
    for (int cidx = lane; cidx < HID; cidx += 32)
        out[rb + cidx] = a0 * x1[rb + cidx] + a1 * x2[rb + cidx] + a2 * x3[rb + cidx];
}

// ---------------------------------------------------------------------------
// Host orchestration
// ---------------------------------------------------------------------------
static inline long cdivl(long a, long b) { return (a + b - 1) / b; }

enum {
    I_X = 0, I_EA = 1, I_EI = 2,
    I_C1_WREL = 3, I_C1_BREL = 4, I_C1_WROOT = 5,
    I_C2_WREL = 6, I_C2_BREL = 7, I_C2_WROOT = 8,
    I_C3_WREL = 9, I_C3_BREL = 10, I_C3_WROOT = 11,
    I_C4_WL = 12, I_C4_BL = 13, I_C4_WR = 14,
    I_C5_WL = 15, I_C5_BL = 16, I_C5_WR = 17,
    I_C6_WL = 18, I_C6_BL = 19, I_C6_WR = 20,
    I_JK1 = 21,   // Wih_f,Whh_f,bih_f,bhh_f,Wih_b,Whh_b,bih_b,bhh_b,Watt,batt
    I_JK2 = 31,
    I_L1W = 41, I_L1B = 42, I_L2W = 43, I_L2B = 44,
    I_A1 = 45, I_A7 = 51
};

extern "C" void kernel_launch(void* const* d_in, const int* in_sizes, int n_in,
                              void* d_out, int out_size, void* d_ws, size_t ws_size,
                              hipStream_t stream)
{
    const float* x   = (const float*)d_in[I_X];
    const float* ea  = (const float*)d_in[I_EA];
    const int*   src = (const int*)d_in[I_EI];
    const int*   dst = src + N_EDGES;
    float* out = (float*)d_out;
    (void)in_sizes; (void)n_in; (void)out_size; (void)ws_size;

    // ---- carve workspace ----
    char* wsp = (char*)d_ws;
    auto carve = [&](size_t bytes) -> char* {
        char* p = wsp; wsp += (bytes + 255) & ~(size_t)255; return p;
    };
    const long NH = (long)N_NODES * HID;
    float* x1    = (float*)carve(NH * 4);
    float* x2    = (float*)carve(NH * 4);
    float* x3    = (float*)carve(NH * 4);
    float* agg   = (float*)carve(NH * 4);
    float* hjk   = (float*)carve(NH * 4);
    float* h63   = (float*)carve((long)N_NODES * 63 * 4);
    float* g     = (float*)carve((long)N_NODES * G4 * 4);
    float* hc    = (float*)carve((long)N_NODES * LSTM_H * 4);
    float* cc    = (float*)carve((long)N_NODES * LSTM_H * 4);
    float* hallf = (float*)carve((long)N_NODES * 3 * LSTM_H * 4);
    float* hallb = (float*)carve((long)N_NODES * 3 * LSTM_H * 4);
    float* bbuf  = (float*)carve(G4 * 4);
    __bf16* abf  = (__bf16*)carve((long)N_NODES * 192 * 2);   // A staging, max KP=192

    auto carve_bf = [&](long n) -> __bf16* { return (__bf16*)carve((size_t)n * 2); };
    // Bt layouts: [Np x Kp] zero-padded
    __bf16* w_c1r  = carve_bf(128L * 32);
    __bf16* w_c1o  = carve_bf(128L * 32);
    __bf16* w_c2r  = carve_bf(128L * 128);
    __bf16* w_c2o  = carve_bf(128L * 128);
    __bf16* w_c3r  = carve_bf(128L * 128);
    __bf16* w_c3o  = carve_bf(128L * 128);
    __bf16* w_c4l  = carve_bf(128L * 128);
    __bf16* w_c4r  = carve_bf(128L * 128);
    __bf16* w_c5l  = carve_bf(128L * 128);
    __bf16* w_c5r  = carve_bf(128L * 128);
    __bf16* w_c6l  = carve_bf(128L * 128);
    __bf16* w_c6r  = carve_bf(128L * 128);
    __bf16* w_ihf1 = carve_bf(704L * 128);
    __bf16* w_hhf1 = carve_bf(704L * 192);
    __bf16* w_ihb1 = carve_bf(704L * 128);
    __bf16* w_hhb1 = carve_bf(704L * 192);
    __bf16* w_ihf2 = carve_bf(704L * 128);
    __bf16* w_hhf2 = carve_bf(704L * 192);
    __bf16* w_ihb2 = carve_bf(704L * 128);
    __bf16* w_hhb2 = carve_bf(704L * 192);
    __bf16* w_l1   = carve_bf(64L * 128);
    __bf16* w_l2   = carve_bf(64L * 64);

    // ---- helpers ----
    auto fill = [&](float* p, float v, long n) {
        fill_kernel<<<(unsigned)cdivl(n, 256), 256, 0, stream>>>(p, v, n);
    };
    auto prep = [&](int idx, int K, int N_, int Kp, int Np, __bf16* dstw, int tr) {
        long n = (long)Kp * Np;
        prep_weight_kernel<<<(unsigned)cdivl(n, 256), 256, 0, stream>>>(
            (const float*)d_in[idx], dstw, K, N_, Kp, Np, tr);
    };
    // Stage A (f32, stride C) into abf (bf16, stride Kp) then run the GEMM.
    auto gemm = [&](const float* Asrc, int C, int Kp, const __bf16* Bt,
                    const float* bias, const float* pa,
                    float* D, int ldd, int N_, int flags) {
        long n = (long)N_NODES * Kp;
        cvt_bf16_kernel<<<(unsigned)cdivl(n, 256), 256, 0, stream>>>(
            Asrc, abf, C, Kp, N_NODES);
        int Np = (int)cdivl(N_, 64) * 64;
        dim3 grid((unsigned)cdivl(cdivl(N_NODES, 16), 4), (unsigned)(Np / 64));
        switch (Kp) {
        case 32:  gemm_wmma_kernel<32><<<grid, 128, 0, stream>>>(abf, Bt, bias, pa, D, ldd, N_NODES, N_, flags); break;
        case 64:  gemm_wmma_kernel<64><<<grid, 128, 0, stream>>>(abf, Bt, bias, pa, D, ldd, N_NODES, N_, flags); break;
        case 128: gemm_wmma_kernel<128><<<grid, 128, 0, stream>>>(abf, Bt, bias, pa, D, ldd, N_NODES, N_, flags); break;
        default:  gemm_wmma_kernel<192><<<grid, 128, 0, stream>>>(abf, Bt, bias, pa, D, ldd, N_NODES, N_, flags); break;
        }
    };
    auto ssum = [&](const float* xin, float* aggb, int C) {
        long n = (long)N_EDGES * C;
        scatter_sum_kernel<<<(unsigned)cdivl(n, 256), 256, 0, stream>>>(
            xin, src, dst, ea, aggb, N_EDGES, C);
    };
    auto smax = [&](const float* xin, float* aggb, int C, int use_ew) {
        long n = (long)N_EDGES * C;
        scatter_max_kernel<<<(unsigned)cdivl(n, 256), 256, 0, stream>>>(
            xin, src, dst, ea, aggb, N_EDGES, C, use_ew);
        finalize_max_kernel<<<(unsigned)cdivl((long)N_NODES * C, 256), 256, 0, stream>>>(
            aggb, (long)N_NODES * C);
    };

    // ---- weight prep (tiny) ----
    prep(I_C1_WREL, D_IN, HID, 32, 128, w_c1r, 1);  prep(I_C1_WROOT, D_IN, HID, 32, 128, w_c1o, 1);
    prep(I_C2_WREL, HID, HID, 128, 128, w_c2r, 1);  prep(I_C2_WROOT, HID, HID, 128, 128, w_c2o, 1);
    prep(I_C3_WREL, HID, HID, 128, 128, w_c3r, 1);  prep(I_C3_WROOT, HID, HID, 128, 128, w_c3o, 1);
    prep(I_C4_WL, HID, HID, 128, 128, w_c4l, 1);    prep(I_C4_WR, HID, HID, 128, 128, w_c4r, 1);
    prep(I_C5_WL, HID, HID, 128, 128, w_c5l, 1);    prep(I_C5_WR, HID, HID, 128, 128, w_c5r, 1);
    prep(I_C6_WL, HID, HID, 128, 128, w_c6l, 1);    prep(I_C6_WR, HID, HID, 128, 128, w_c6r, 1);
    prep(I_JK1 + 0, HID, G4, 128, 704, w_ihf1, 0);  prep(I_JK1 + 1, LSTM_H, G4, 192, 704, w_hhf1, 0);
    prep(I_JK1 + 4, HID, G4, 128, 704, w_ihb1, 0);  prep(I_JK1 + 5, LSTM_H, G4, 192, 704, w_hhb1, 0);
    prep(I_JK2 + 0, HID, G4, 128, 704, w_ihf2, 0);  prep(I_JK2 + 1, LSTM_H, G4, 192, 704, w_hhf2, 0);
    prep(I_JK2 + 4, HID, G4, 128, 704, w_ihb2, 0);  prep(I_JK2 + 5, LSTM_H, G4, 192, 704, w_hhb2, 0);
    prep(I_L1W, HID, 63, 128, 64, w_l1, 1);         prep(I_L2W, 63, 3, 64, 64, w_l2, 1);

    const float* A1 = (const float*)d_in[I_A1 + 0];
    const float* A2 = (const float*)d_in[I_A1 + 1];
    const float* A3 = (const float*)d_in[I_A1 + 2];
    const float* A4 = (const float*)d_in[I_A1 + 3];
    const float* A5 = (const float*)d_in[I_A1 + 4];
    const float* A6 = (const float*)d_in[I_A1 + 5];
    const float* A7 = (const float*)d_in[I_A7];

    // ---- c1 (add aggr, D_IN channels) ----
    fill(agg, 0.0f, (long)N_NODES * D_IN);
    ssum(x, agg, D_IN);
    gemm(agg, D_IN, 32, w_c1r, (const float*)d_in[I_C1_BREL], nullptr, x1, HID, HID, 0);
    gemm(x,   D_IN, 32, w_c1o, nullptr, A1, x1, HID, HID, GF_ACC | GF_PRELU);

    // ---- c2 (add) ----
    fill(agg, 0.0f, NH);
    ssum(x1, agg, HID);
    gemm(agg, HID, 128, w_c2r, (const float*)d_in[I_C2_BREL], nullptr, x2, HID, HID, 0);
    gemm(x1,  HID, 128, w_c2o, nullptr, A2, x2, HID, HID, GF_ACC | GF_PRELU);

    // ---- c3 (max, with edge weight) ----
    fill(agg, -INFINITY, NH);
    smax(x2, agg, HID, 1);
    gemm(agg, HID, 128, w_c3r, (const float*)d_in[I_C3_BREL], nullptr, x3, HID, HID, 0);
    gemm(x2,  HID, 128, w_c3o, nullptr, A3, x3, HID, HID, GF_ACC | GF_PRELU);

    // ---- JK-LSTM runner ----
    auto run_jk = [&](int base, const __bf16* wihf, const __bf16* whhf,
                      const __bf16* wihb, const __bf16* whhb,
                      const float* xs0, const float* xs1, const float* xs2,
                      float* outb) {
        const float* xs[3] = { xs0, xs1, xs2 };
        long nlh = (long)N_NODES * LSTM_H;
        // forward
        vec_add_kernel<<<(unsigned)cdivl(G4, 256), 256, 0, stream>>>(
            (const float*)d_in[base + 2], (const float*)d_in[base + 3], bbuf, G4);
        fill(hc, 0.0f, nlh); fill(cc, 0.0f, nlh);
        for (int t = 0; t < 3; ++t) {
            gemm(xs[t], HID, 128, wihf, bbuf, nullptr, g, G4, G4, 0);
            if (t > 0) gemm(hc, LSTM_H, 192, whhf, nullptr, nullptr, g, G4, G4, GF_ACC);
            lstm_point_kernel<<<(unsigned)cdivl(nlh, 256), 256, 0, stream>>>(
                g, cc, hc, hallf, t, N_NODES);
        }
        // backward
        vec_add_kernel<<<(unsigned)cdivl(G4, 256), 256, 0, stream>>>(
            (const float*)d_in[base + 6], (const float*)d_in[base + 7], bbuf, G4);
        fill(hc, 0.0f, nlh); fill(cc, 0.0f, nlh);
        for (int s = 0; s < 3; ++s) {
            int t = 2 - s;
            gemm(xs[t], HID, 128, wihb, bbuf, nullptr, g, G4, G4, 0);
            if (s > 0) gemm(hc, LSTM_H, 192, whhb, nullptr, nullptr, g, G4, G4, GF_ACC);
            lstm_point_kernel<<<(unsigned)cdivl(nlh, 256), 256, 0, stream>>>(
                g, cc, hc, hallb, t, N_NODES);
        }
        jk_attn_kernel<<<(unsigned)cdivl(N_NODES, 4), 128, 0, stream>>>(
            xs0, xs1, xs2, hallf, hallb,
            (const float*)d_in[base + 8], (const float*)d_in[base + 9],
            outb, N_NODES);
    };

    run_jk(I_JK1, w_ihf1, w_hhf1, w_ihb1, w_hhb1, x1, x2, x3, hjk);

    // ---- SAGE-max c4..c6 (reuse x1..x3 as x4..x6) ----
    fill(agg, -INFINITY, NH);
    smax(hjk, agg, HID, 0);
    gemm(agg, HID, 128, w_c4l, (const float*)d_in[I_C4_BL], nullptr, x1, HID, HID, 0);
    gemm(hjk, HID, 128, w_c4r, nullptr, A4, x1, HID, HID, GF_ACC | GF_PRELU);

    fill(agg, -INFINITY, NH);
    smax(x1, agg, HID, 0);
    gemm(agg, HID, 128, w_c5l, (const float*)d_in[I_C5_BL], nullptr, x2, HID, HID, 0);
    gemm(x1,  HID, 128, w_c5r, nullptr, A5, x2, HID, HID, GF_ACC | GF_PRELU);

    fill(agg, -INFINITY, NH);
    smax(x2, agg, HID, 0);
    gemm(agg, HID, 128, w_c6l, (const float*)d_in[I_C6_BL], nullptr, x3, HID, HID, 0);
    gemm(x2,  HID, 128, w_c6r, nullptr, A6, x3, HID, HID, GF_ACC | GF_PRELU);

    run_jk(I_JK2, w_ihf2, w_hhf2, w_ihb2, w_hhb2, x1, x2, x3, hjk);

    // ---- heads ----
    gemm(hjk, HID, 128, w_l1, (const float*)d_in[I_L1B], A7, h63, 63, 63, GF_PRELU);
    gemm(h63, 63, 64, w_l2, (const float*)d_in[I_L2B], nullptr, out, 3, 3, 0);
}
